// AttNHPTransformer_91225105367374
// MI455X (gfx1250) — matrix operand order repaired and checked by reading, not compiled, not run on hardware
//
#include <hip/hip_runtime.h>
#include <hip/hip_bf16.h>

// ---------------------------------------------------------------------------
// AttNHP transformer forward for MI455X (gfx1250, wave32, WMMA).
// All GEMMs use v_wmma_f32_16x16x32_f16 (f32 accumulate). Attention is a
// flash-style streaming kernel (online softmax, no LxL score tensor).
// Inner loops are software-pipelined (register double-buffering) so fragment
// loads overlap WMMA issue instead of serializing on s_wait_loadcnt 0.
// ---------------------------------------------------------------------------

typedef __attribute__((ext_vector_type(16))) _Float16 v16h;
typedef __attribute__((ext_vector_type(8)))  _Float16 v8h;
typedef __attribute__((ext_vector_type(8)))  float    v8f;

#define BQ    4
#define LQ    1024
#define DINQ  256
#define DMQ   512
#define HQ    8
#define DVQ   64
#define DFFQ  2048
#define NLQ   2
#define PPQ   (2 * HQ * DMQ + HQ * DVQ)   // 8704 = qkv proj width
#define MROWS (BQ * LQ)                   // 4096

// ---------------------------------------------------------------------------
// WMMA fragment loaders (wave32; A and B share this code's contiguous k order)
// A (16x32 f16, MxK): lane m = lane&15, half = lane>>4
//   vec[i] = A[m][k0 + half*16 + i]           (contiguous 32B)
// B (32x16 f16, KxN) built from row-major W[N][K]: lane n = lane&15
//   vec[i] = W[n][k0 + half*16 + i]           (contiguous 32B)
// C/D (16x16 f32): lane n = lane&15, elem r -> row m = (lane>>4)*8 + r
// ---------------------------------------------------------------------------

__device__ __forceinline__ v16h frag_a(const _Float16* __restrict__ base,
                                       int pitch, int row0, int kbase) {
  const int lane = threadIdx.x & 31;
  const int lh   = lane >> 4;
  const _Float16* p = base + (size_t)(row0 + (lane & 15)) * pitch + kbase + lh * 16;
  return *(const v16h*)p;
}

__device__ __forceinline__ v16h frag_b(const _Float16* __restrict__ base,
                                       int pitch, int col0, int kbase) {
  const int lane = threadIdx.x & 31;
  const _Float16* p = base + (size_t)(col0 + (lane & 15)) * pitch + kbase + (lane >> 4) * 16;
  return *(const v16h*)p;
}

#define WMMA_F16(A_, B_, C_) \
  __builtin_amdgcn_wmma_f32_16x16x32_f16(false, (A_), false, (B_), (short)0, (C_), false, false)

// ---------------------------------------------------------------------------
// Generic GEMM:  C[M,N] = A[M,K] @ W[N,K]^T + bias[N]   (+optional ReLU)
// Block = 256 threads (8 waves). Tile: 256 rows x 64 cols.
// Wave w owns rows m0..m0+31 (two 16-row WMMA tiles) x 4 n-tiles of 16:
// 8 WMMAs per 32-wide k-step, B-fragments reused across both M-tiles.
// One k-step of register double-buffering hides load latency under WMMA.
// ---------------------------------------------------------------------------
template <int OUT_HALF, int RELU>
__global__ __launch_bounds__(256) void k_gemm(const _Float16* __restrict__ A,
                                              const _Float16* __restrict__ W,
                                              const float* __restrict__ bias,
                                              float* __restrict__ Cf,
                                              _Float16* __restrict__ Ch,
                                              int N, int K) {
  const int lane = threadIdx.x & 31;
  const int wave = threadIdx.x >> 5;
  const int m0 = blockIdx.y * 256 + wave * 32;
  const int n0 = blockIdx.x * 64;
  const int nl = lane & 15;
  const int lh = lane >> 4;

  v8f acc[2][4] = {};

  // prologue: fragments for k = 0
  v16h a0 = frag_a(A, K, m0, 0);
  v16h a1 = frag_a(A, K, m0 + 16, 0);
  v16h bf[4];
#pragma unroll
  for (int t = 0; t < 4; ++t) bf[t] = frag_b(W, K, n0 + t * 16, 0);

  for (int k = 0; k + 32 < K; k += 32) {
    // issue next k-step's loads before consuming current fragments
    if (k + 64 < K)  // pull the k+64 A strip toward L2 (global_prefetch_b8)
      __builtin_prefetch(A + (size_t)(m0 + nl) * K + k + 64, 0, 1);
    v16h a0n = frag_a(A, K, m0, k + 32);
    v16h a1n = frag_a(A, K, m0 + 16, k + 32);
    v16h bfn[4];
#pragma unroll
    for (int t = 0; t < 4; ++t) bfn[t] = frag_b(W, K, n0 + t * 16, k + 32);

#pragma unroll
    for (int t = 0; t < 4; ++t) {
      acc[0][t] = WMMA_F16(a0, bf[t], acc[0][t]);
      acc[1][t] = WMMA_F16(a1, bf[t], acc[1][t]);
    }
    a0 = a0n; a1 = a1n;
#pragma unroll
    for (int t = 0; t < 4; ++t) bf[t] = bfn[t];
  }
  // epilogue k-step
#pragma unroll
  for (int t = 0; t < 4; ++t) {
    acc[0][t] = WMMA_F16(a0, bf[t], acc[0][t]);
    acc[1][t] = WMMA_F16(a1, bf[t], acc[1][t]);
  }

#pragma unroll
  for (int mi = 0; mi < 2; ++mi) {
#pragma unroll
    for (int t = 0; t < 4; ++t) {
      const int   n  = n0 + t * 16 + nl;
      const float bv = bias[n];
#pragma unroll
      for (int r = 0; r < 8; ++r) {
        const int m = m0 + mi * 16 + lh * 8 + r;
        float v = acc[mi][t][r] + bv;
        if (RELU) v = fmaxf(v, 0.0f);
        if (OUT_HALF) Ch[(size_t)m * N + n] = (_Float16)v;
        else          Cf[(size_t)m * N + n] = v;
      }
    }
  }
}

// ---------------------------------------------------------------------------
// V^T fragment: A-operand of O^T = V^T @ P^T; strided 16-bit gathers.
// ---------------------------------------------------------------------------
__device__ __forceinline__ v16h frag_vt(const _Float16* __restrict__ vp, int kc, int t2) {
  const int lane = threadIdx.x & 31;
  const int nl = lane & 15, lh = lane >> 4;
  const int dv = t2 * 16 + nl;
  v16h r;
#pragma unroll
  for (int i = 0; i < 8; ++i) {
    r[i]     = vp[(size_t)(kc + lh * 16 + i) * PPQ + dv];
    r[8 + i] = vp[(size_t)(kc + lh * 16 + 8 + i) * PPQ + dv];
  }
  return r;
}

// ---------------------------------------------------------------------------
// Flash attention per (b, h, 16-query tile). One wave per query tile.
// S^T = K_tile(16x512) @ Q^T : D-layout lane = query, vgpr = key  ->
// online softmax is (mostly) lane-local; one xor-16 merges half-waves.
// O^T = V^T(16dv x 32key) @ P^T(32key x 16q), accumulated f32, rescaled.
// Causal-strict mask (query <= key) + padding (key >= seq_len) as -1e6,
// exp underflow reproduces the reference's exact zero weights.
// ---------------------------------------------------------------------------
__global__ __launch_bounds__(256) void k_attn(const _Float16* __restrict__ proj,
                                              const int* __restrict__ seql,
                                              _Float16* __restrict__ sa) {
  const int lane = threadIdx.x & 31;
  const int wave = threadIdx.x >> 5;
  const int b = blockIdx.z;
  const int h = blockIdx.y;
  const int q0 = (blockIdx.x * 8 + wave) * 16;
  const int nl = lane & 15;
  const int lh = lane >> 4;

  const size_t rowbase = (size_t)b * LQ * PPQ;
  const _Float16* qp = proj + rowbase + (size_t)h * DMQ;
  const _Float16* kp = proj + rowbase + (size_t)(HQ * DMQ) + (size_t)h * DMQ;
  const _Float16* vp = proj + rowbase + (size_t)(2 * HQ * DMQ) + (size_t)h * DVQ;
  const int sl = seql[b];
  const int query = q0 + nl;

  // Preload the 16 Q B-fragments (reused by every key tile): 128 VGPRs.
  v16h qf[16];
#pragma unroll
  for (int kk = 0; kk < 16; ++kk)
    qf[kk] = *(const v16h*)(qp + (size_t)query * PPQ + kk * 32 + lh * 16);

  v8f o[4] = {};
  float runmax = -3.0e38f, runsum = 0.0f;
  const float scale = 0.044194173824159216f;  // 1/sqrt(512)

  int kend = q0 + 15; if (sl < kend) kend = sl;   // exclusive key bound
  if (kend < 1) kend = 1;
  const int nchunk = (kend + 31) >> 5;

  for (int c = 0; c < nchunk; ++c) {
    const int kc = c * 32;
    // ---- scores S^T for 32 keys (two 16-key tiles), K-dim d = 512 ----
    // Pipelined: next K-row fragments issued before current WMMAs.
    v8f s0 = {}, s1 = {};
    v16h ka0 = frag_a(kp, PPQ, kc, 0);
    v16h ka1 = frag_a(kp, PPQ, kc + 16, 0);
#pragma unroll
    for (int kk = 0; kk < 15; ++kk) {
      v16h na0 = frag_a(kp, PPQ, kc, (kk + 1) * 32);
      v16h na1 = frag_a(kp, PPQ, kc + 16, (kk + 1) * 32);
      s0 = WMMA_F16(ka0, qf[kk], s0);
      s1 = WMMA_F16(ka1, qf[kk], s1);
      ka0 = na0; ka1 = na1;
    }
    s0 = WMMA_F16(ka0, qf[15], s0);
    s1 = WMMA_F16(ka1, qf[15], s1);

    // Issue the first V^T gather now; it completes under the softmax VALU work.
    v16h vt = frag_vt(vp, kc, 0);

    // ---- scale + mask + online softmax ----
    float p0[8], p1[8];
    float tmax = -3.0e38f;
#pragma unroll
    for (int r = 0; r < 8; ++r) {
      const int key0 = kc + lh * 8 + r;
      const int key1 = kc + 16 + lh * 8 + r;
      float v0 = s0[r] * scale;
      float v1 = s1[r] * scale;
      if (key0 >= sl || query <= key0) v0 = -1.0e6f;
      if (key1 >= sl || query <= key1) v1 = -1.0e6f;
      p0[r] = v0; p1[r] = v1;
      tmax = fmaxf(tmax, fmaxf(v0, v1));
    }
    tmax = fmaxf(tmax, __shfl_xor(tmax, 16, 32));
    const float newmax = fmaxf(runmax, tmax);
    const float corr   = __expf(runmax - newmax);
    runmax = newmax;
    float psum = 0.0f;
#pragma unroll
    for (int r = 0; r < 8; ++r) {
      p0[r] = __expf(p0[r] - newmax);
      p1[r] = __expf(p1[r] - newmax);
      psum += p0[r] + p1[r];
    }
    psum += __shfl_xor(psum, 16, 32);
    runsum = runsum * corr + psum;
#pragma unroll
    for (int t2 = 0; t2 < 4; ++t2)
#pragma unroll
      for (int r = 0; r < 8; ++r) o[t2][r] *= corr;

    // ---- build P^T B-fragment (32 keys x 16 queries) via xor-16 swaps ----
    v16h pb;
#pragma unroll
    for (int r = 0; r < 8; ++r) {
      const float o0 = __shfl_xor(p0[r], 16, 32);
      const float o1 = __shfl_xor(p1[r], 16, 32);
      const float lo = (lh == 0) ? p0[r] : o1;  // keys half*16 + 0..7
      const float hi = (lh == 0) ? o0 : p1[r];  // keys half*16 + 8..15
      pb[r]     = (_Float16)lo;
      pb[8 + r] = (_Float16)hi;
    }

    // ---- O^T += V^T @ P^T over the 4 dv tiles (pipelined gathers) ----
#pragma unroll
    for (int t2 = 0; t2 < 3; ++t2) {
      v16h vn = frag_vt(vp, kc, t2 + 1);
      o[t2] = WMMA_F16(vt, pb, o[t2]);
      vt = vn;
    }
    o[3] = WMMA_F16(vt, pb, o[3]);
  }

  // ---- normalize and store sa[b, q, h*64 + dv] as f16 ----
  const float inv = 1.0f / runsum;
#pragma unroll
  for (int t2 = 0; t2 < 4; ++t2) {
    v8h pk;
#pragma unroll
    for (int r = 0; r < 8; ++r) pk[r] = (_Float16)(o[t2][r] * inv);
    const size_t addr = ((size_t)b * LQ + query) * (HQ * DVQ) + h * DVQ + t2 * 16 + lh * 8;
    *(v8h*)(sa + addr) = pk;
  }
}

// ---------------------------------------------------------------------------
// Elementwise / reduction helpers
// ---------------------------------------------------------------------------
__global__ void k_cast_f16(const float* __restrict__ src, _Float16* __restrict__ dst, int n) {
  const int i = blockIdx.x * 256 + threadIdx.x;
  if (i < n) dst[i] = (_Float16)src[i];
}

__global__ void k_posenc(float* __restrict__ x, _Float16* __restrict__ xh,
                         const float* __restrict__ times) {
  const int idx = blockIdx.x * 256 + threadIdx.x;   // over MROWS*DM
  const int j   = idx & (DMQ - 1);
  const int row = idx >> 9;
  const float t  = times[row];
  const int   jj = (j < DMQ / 2) ? j : (j - DMQ / 2);
  const float d  = (float)jj * (1.0f / 255.0f);
  const float denum = __expf(d * 9.210340371976184f);  // (5*M_HI/M_LO)^d = 10000^d
  const float arg = t / denum;
  const float pe  = (j < DMQ / 2) ? sinf(arg) : cosf(arg);
  const float v = x[idx] + pe;
  x[idx]  = v;
  xh[idx] = (_Float16)v;
}

// x_out = LayerNorm(x_in + (invalid ? 0 : add)) * g + be ; also emits f16 copy.
__global__ __launch_bounds__(256) void k_residual_ln(const float* __restrict__ xin,
                                                     const float* __restrict__ addv,
                                                     const int* __restrict__ seql,
                                                     const float* __restrict__ g,
                                                     const float* __restrict__ be,
                                                     float* __restrict__ xout,
                                                     _Float16* __restrict__ xouth,
                                                     int zero_invalid) {
  const int row = blockIdx.x;
  const int b = row >> 10;          // row / L
  const int l = row & (LQ - 1);
  bool invalid = false;
  if (zero_invalid) { const int sl = seql[b]; invalid = (l == 0) || (sl <= 0); }

  const size_t base = (size_t)row * DMQ;
  const int t = threadIdx.x;
  const int j0 = t, j1 = t + 256;
  const float a0 = invalid ? 0.0f : addv[base + j0];
  const float a1 = invalid ? 0.0f : addv[base + j1];
  const float v0 = xin[base + j0] + a0;
  const float v1 = xin[base + j1] + a1;
  float s = v0 + v1, s2 = v0 * v0 + v1 * v1;
#pragma unroll
  for (int off = 16; off >= 1; off >>= 1) {
    s  += __shfl_xor(s, off, 32);
    s2 += __shfl_xor(s2, off, 32);
  }
  __shared__ float r0[8], r1[8];
  const int w = t >> 5;
  if ((t & 31) == 0) { r0[w] = s; r1[w] = s2; }
  __syncthreads();
  s = 0.0f; s2 = 0.0f;
#pragma unroll
  for (int i = 0; i < 8; ++i) { s += r0[i]; s2 += r1[i]; }
  const float mu  = s * (1.0f / DMQ);
  const float var = s2 * (1.0f / DMQ) - mu * mu;
  const float inv = rsqrtf(var + 1e-5f);
  const float o0 = (v0 - mu) * inv * g[j0] + be[j0];
  const float o1 = (v1 - mu) * inv * g[j1] + be[j1];
  xout[base + j0] = o0;          xout[base + j1] = o1;
  xouth[base + j0] = (_Float16)o0; xouth[base + j1] = (_Float16)o1;
}

// ---------------------------------------------------------------------------
// Orchestration
// ---------------------------------------------------------------------------
extern "C" void kernel_launch(void* const* d_in, const int* in_sizes, int n_in,
                              void* d_out, int out_size, void* d_ws, size_t ws_size,
                              hipStream_t stream) {
  (void)in_sizes; (void)n_in; (void)out_size; (void)ws_size;

  const float* emb   = (const float*)d_in[0];
  const float* times = (const float*)d_in[1];
  const int*   seql  = (const int*)d_in[2];
  const float* Wp    = (const float*)d_in[3];
  const float* bp    = (const float*)d_in[4];
  const float* Win   = (const float*)d_in[5];
  const float* bin_  = (const float*)d_in[6];
  const float* Wout  = (const float*)d_in[7];
  const float* bout  = (const float*)d_in[8];
  const float* g1    = (const float*)d_in[9];
  const float* be1   = (const float*)d_in[10];
  const float* W1    = (const float*)d_in[11];
  const float* b1    = (const float*)d_in[12];
  const float* W2    = (const float*)d_in[13];
  const float* b2    = (const float*)d_in[14];
  const float* g2    = (const float*)d_in[15];
  const float* be2   = (const float*)d_in[16];
  float* out = (float*)d_out;

  // ---- workspace carve (256B aligned) ----
  char* p = (char*)d_ws;
  auto take = [&](size_t bytes) -> void* {
    void* r = (void*)p;
    p += (bytes + 255) & ~(size_t)255;
    return r;
  };
  _Float16* emb_h  = (_Float16*)take((size_t)MROWS * DINQ * 2);
  _Float16* Wp_h   = (_Float16*)take((size_t)DMQ * DINQ * 2);
  _Float16* Win_h  = (_Float16*)take((size_t)NLQ * PPQ * DMQ * 2);
  _Float16* Wout_h = (_Float16*)take((size_t)NLQ * DMQ * (HQ * DVQ) * 2);
  _Float16* W1_h   = (_Float16*)take((size_t)NLQ * DFFQ * DMQ * 2);
  _Float16* W2_h   = (_Float16*)take((size_t)NLQ * DMQ * DFFQ * 2);
  float*    x_f    = (float*)take((size_t)MROWS * DMQ * 4);
  _Float16* x_h    = (_Float16*)take((size_t)MROWS * DMQ * 2);
  float*    tmp_f  = (float*)take((size_t)MROWS * DMQ * 4);
  _Float16* sa_h   = (_Float16*)take((size_t)MROWS * DMQ * 2);
  _Float16* proj_h = (_Float16*)take((size_t)MROWS * PPQ * 2);
  _Float16* ff1_h  = proj_h;  // proj dead after attention; reuse for FFN hidden

  auto cg = [](int n) { return dim3((unsigned)((n + 255) / 256)); };

  // ---- f32 -> f16 copies of activations-in and weights ----
  k_cast_f16<<<cg(MROWS * DINQ), 256, 0, stream>>>(emb,  emb_h,  MROWS * DINQ);
  k_cast_f16<<<cg(DMQ * DINQ),   256, 0, stream>>>(Wp,   Wp_h,   DMQ * DINQ);
  k_cast_f16<<<cg(NLQ * PPQ * DMQ), 256, 0, stream>>>(Win, Win_h, NLQ * PPQ * DMQ);
  k_cast_f16<<<cg(NLQ * DMQ * HQ * DVQ), 256, 0, stream>>>(Wout, Wout_h, NLQ * DMQ * HQ * DVQ);
  k_cast_f16<<<cg(NLQ * DFFQ * DMQ), 256, 0, stream>>>(W1, W1_h, NLQ * DFFQ * DMQ);
  k_cast_f16<<<cg(NLQ * DMQ * DFFQ), 256, 0, stream>>>(W2, W2_h, NLQ * DMQ * DFFQ);

  const dim3 blk(256);
  const unsigned gy = MROWS / 256;  // 16 (block tile = 256 rows)

  // ---- x = emb @ Wp^T + bp, then += positional encoding ----
  k_gemm<0, 0><<<dim3(DMQ / 64, gy), blk, 0, stream>>>(emb_h, Wp_h, bp, x_f, nullptr,
                                                       DMQ, DINQ);
  k_posenc<<<cg(MROWS * DMQ), 256, 0, stream>>>(x_f, x_h, times);

  // ---- transformer layers ----
  for (int l = 0; l < NLQ; ++l) {
    // qkv projection -> f16
    k_gemm<1, 0><<<dim3(PPQ / 64, gy), blk, 0, stream>>>(
        x_h, Win_h + (size_t)l * PPQ * DMQ, bin_ + (size_t)l * PPQ,
        nullptr, proj_h, PPQ, DMQ);
    // flash attention -> sa (f16)
    k_attn<<<dim3(LQ / 16 / 8, HQ, BQ), blk, 0, stream>>>(proj_h, seql, sa_h);
    // sa @ Wout^T + bout -> tmp (f32)
    k_gemm<0, 0><<<dim3(DMQ / 64, gy), blk, 0, stream>>>(
        sa_h, Wout_h + (size_t)l * DMQ * HQ * DVQ, bout + (size_t)l * DMQ,
        tmp_f, nullptr, DMQ, HQ * DVQ);
    // x = LN(x + where(invalid,0,sa_out)) * g1 + be1
    k_residual_ln<<<MROWS, blk, 0, stream>>>(x_f, tmp_f, seql,
                                             g1 + (size_t)l * DMQ, be1 + (size_t)l * DMQ,
                                             x_f, x_h, 1);
    // ff1 = relu(x @ W1^T + b1) -> f16 (reuses proj buffer)
    k_gemm<1, 1><<<dim3(DFFQ / 64, gy), blk, 0, stream>>>(
        x_h, W1_h + (size_t)l * DFFQ * DMQ, b1 + (size_t)l * DFFQ,
        nullptr, ff1_h, DFFQ, DMQ);
    // ff2 = ff1 @ W2^T + b2 -> f32
    k_gemm<0, 0><<<dim3(DMQ / 64, gy), blk, 0, stream>>>(
        ff1_h, W2_h + (size_t)l * DMQ * DFFQ, b2 + (size_t)l * DMQ,
        tmp_f, nullptr, DMQ, DFFQ);
    // x = LN(x + ff2) * g2 + be2 ; last layer writes straight to d_out
    float* xo = (l == NLQ - 1) ? out : x_f;
    k_residual_ln<<<MROWS, blk, 0, stream>>>(x_f, tmp_f, nullptr,
                                             g2 + (size_t)l * DMQ, be2 + (size_t)l * DMQ,
                                             xo, x_h, 0);
  }
}
